// DeformConv2d_67199058313461
// MI455X (gfx1250) — matrix operand order repaired and checked by reading, not compiled, not run on hardware
//
#include <hip/hip_runtime.h>
#include <hip/hip_bf16.h>

typedef __attribute__((ext_vector_type(16))) _Float16 v16h;
typedef __attribute__((ext_vector_type(8)))  float    v8f;

namespace {

constexpr int N_ = 8, C_ = 64, H_ = 128, W_ = 512, CO_ = 64;
constexpr int PAD = 1;
constexpr int TILE = 64;       // pixels per block
constexpr int THREADS = 256;   // 8 wave32 waves

constexpr int WS_OFF_ELEMS = 9 * 32 * 64;   // f16, rows >=18 zero-padded
constexpr int WS_DEF_ELEMS = 9 * 64 * 64;   // f16
constexpr size_t WS_NEEDED = (size_t)(WS_OFF_ELEMS + WS_DEF_ELEMS) * sizeof(_Float16);

union FragH  { v16h h; uint4 q[2]; };
union AccF   { v8f  v; float f[8]; };
union Pack16 { _Float16 h[16]; uint4 q[2]; };

// A-matrix (16x32 f16) fragment from row-major [rows][64] f16 LDS tile.
// lanes 0-15 -> M=lane,    v0..3 = K 0..7,  v4..7 = K 16..23
// lanes16-31 -> M=lane-16, v0..3 = K 8..15, v4..7 = K 24..31
__device__ inline v16h load_a(const _Float16* base, int lane, int kk) {
  const int row = lane & 15;
  const int kh  = (lane >> 4) << 3;
  const _Float16* r = base + row * 64 + kk * 32 + kh;
  FragH f;
  f.q[0] = *reinterpret_cast<const uint4*>(r);
  f.q[1] = *reinterpret_cast<const uint4*>(r + 16);
  return f.h;
}

// B-matrix (32x16 f16) fragment from sB[px][ch] LDS tile (K contiguous per row).
// lanes 0-15 -> N=lane,    v0..3 = K 0..7,  v4..7 = K 16..23
// lanes16-31 -> N=lane-16, v0..3 = K 8..15, v4..7 = K 24..31
__device__ inline v16h load_b(const _Float16 (*sB)[64], int lane, int kk) {
  const int n   = lane & 15;
  const int sec = (lane >> 4) << 3;
  const _Float16* r = &sB[n][kk * 32 + sec];
  FragH f;
  f.q[0] = *reinterpret_cast<const uint4*>(r);
  f.q[1] = *reinterpret_cast<const uint4*>(r + 16);
  return f.h;
}

// ---- one-shot weight transpose: global f32 [o][c][3][3] -> ws f16 [t][o][c] ----
__global__ __launch_bounds__(THREADS)
void prep_weights(const float* __restrict__ w_off,
                  const float* __restrict__ w_def,
                  _Float16* __restrict__ ws) {
  const int e = blockIdx.x * THREADS + threadIdx.x;
  _Float16* wsOff = ws;
  _Float16* wsDef = ws + WS_OFF_ELEMS;
  if (e < WS_OFF_ELEMS) {
    const int t = e / (32 * 64);
    const int r = e % (32 * 64);
    const int o = r >> 6, c = r & 63;
    wsOff[e] = (o < 18) ? (_Float16)w_off[(o * 64 + c) * 9 + t] : (_Float16)0.0f;
  }
  if (e < WS_DEF_ELEMS) {
    const int t = e / 4096;
    const int r = e % 4096;
    const int o = r >> 6, c = r & 63;
    wsDef[e] = (_Float16)w_def[(o * 64 + c) * 9 + t];
  }
}

template <bool USE_WS>
__global__ __launch_bounds__(THREADS)
void deform_conv_fused(const float* __restrict__ x,
                       const float* __restrict__ w_off,
                       const float* __restrict__ b_off,
                       const float* __restrict__ w_def,
                       const float* __restrict__ b_def,
                       const _Float16* __restrict__ ws,
                       float* __restrict__ out) {
  __shared__ __align__(16) _Float16 sW[64][64];   // tap weights (f16)
  __shared__ __align__(16) _Float16 sB[64][64];   // [px][ch] activation tile (f16)
  __shared__ __align__(16) float    sOff[32][64]; // rows 0..17 real, 18..31 scratch
  __shared__ float sBoff[32];                     // b_off zero-padded
  __shared__ float sBdef[64];                     // b_def

  const int tid  = threadIdx.x;
  const int lane = tid & 31;
  const int wv   = tid >> 5;

  const int x0 = blockIdx.x * TILE;
  const int y  = blockIdx.y;
  const int n  = blockIdx.z;

  const int p  = tid & 63;          // pixel owned in builders
  const int cb = (tid >> 6) << 4;   // 16-channel slab owned in builders

  const float* xb = x + (size_t)n * C_ * H_ * W_;
  const _Float16* wsOff = ws;
  const _Float16* wsDef = ws + WS_OFF_ELEMS;

  // one-time bias preload into LDS (consumed only after several barriers)
  if (tid < 32) sBoff[tid] = (tid < 18) ? b_off[tid] : 0.0f;
  if (tid >= 64 && tid < 128) sBdef[tid - 64] = b_def[tid - 64];

  // ================= Stage A: offset-predicting conv (64 -> 18 ch) =================
  {
    const int mtA = wv >> 2;        // 0..1
    const int ntA = wv & 3;         // 0..3
    AccF acc;
    #pragma unroll
    for (int r = 0; r < 8; ++r) acc.f[r] = 0.0f;

    for (int t = 0; t < 9; ++t) {
      const int di = t / 3, dj = t % 3;
      // stage tap weights [32][64]
      if constexpr (USE_WS) {
        const uint4* src = reinterpret_cast<const uint4*>(wsOff + t * 32 * 64);
        reinterpret_cast<uint4*>(&sW[0][0])[tid] = src[tid];   // 256 x 16B
      } else {
        for (int e = tid; e < 32 * 64; e += THREADS) {
          const int o = e >> 6, c = e & 63;
          const float v = (o < 18) ? w_off[(o * 64 + c) * 9 + t] : 0.0f;
          sW[o][c] = (_Float16)v;
        }
      }
      // build B tile: sB[p][c] = x[n][c][y+di-1][x0+p+dj-1] (zero outside).
      // Unconditional loads from a clamped address + 0/1 mask -> no per-load branches.
      {
        const int yy = y + di - PAD;
        const int xx = x0 + p + dj - PAD;
        const float m = ((yy >= 0) && (yy < H_) && (xx >= 0) && (xx < W_)) ? 1.0f : 0.0f;
        const int yyc = min(max(yy, 0), H_ - 1);
        const int xxc = min(max(xx, 0), W_ - 1);
        const float* pc = xb + (size_t)cb * H_ * W_ + yyc * W_ + xxc;
        float vals[16];
        #pragma unroll
        for (int cc = 0; cc < 16; ++cc) vals[cc] = pc[(size_t)cc * H_ * W_];
        Pack16 pk;
        #pragma unroll
        for (int cc = 0; cc < 16; ++cc) pk.h[cc] = (_Float16)(m * vals[cc]);
        *reinterpret_cast<uint4*>(&sB[p][cb])     = pk.q[0];
        *reinterpret_cast<uint4*>(&sB[p][cb + 8]) = pk.q[1];
      }
      __syncthreads();
      #pragma unroll
      for (int kk = 0; kk < 2; ++kk) {
        const v16h a = load_a(&sW[mtA * 16][0], lane, kk);
        const v16h b = load_b(sB, lane, kk);
        acc.v = __builtin_amdgcn_wmma_f32_16x16x32_f16(
            false, a, false, b, (short)0, acc.v, false, false);
      }
      __syncthreads();
    }
    // scatter D-tile (+ b_off) into sOff -- unconditional (rows 18..31 are scratch)
    #pragma unroll
    for (int r = 0; r < 8; ++r) {
      const int o  = mtA * 16 + r + ((lane >> 4) << 3);   // 0..31
      const int pp = ntA * 16 + (lane & 15);
      sOff[o][pp] = acc.f[r] + sBoff[o];
    }
  }
  __syncthreads();

  // ================= Stage B: deformable conv (bilinear sample + 64x64 GEMM) =======
  const int mt = wv >> 2;   // wave owns tiles (mt, nt) and (mt+2, nt)
  const int nt = wv & 3;
  AccF acc0, acc1;
  #pragma unroll
  for (int r = 0; r < 8; ++r) { acc0.f[r] = 0.0f; acc1.f[r] = 0.0f; }

  for (int t = 0; t < 9; ++t) {
    const int di = t / 3, dj = t % 3;
    // stage tap weights [64][64]
    if constexpr (USE_WS) {
      const uint4* src = reinterpret_cast<const uint4*>(wsDef + t * 64 * 64);
      uint4* dst = reinterpret_cast<uint4*>(&sW[0][0]);
      dst[tid]       = src[tid];          // 512 x 16B total
      dst[tid + 256] = src[tid + 256];
    } else {
      for (int e = tid; e < 64 * 64; e += THREADS) {
        const int o = e >> 6, c = e & 63;
        sW[o][c] = (_Float16)w_def[(o * 64 + c) * 9 + t];
      }
    }
    // bilinear-sampled B tile (coords/weights computed once per pixel, reused over 16 ch)
    {
      const float dy = sOff[2 * t + 0][p];
      const float dx = sOff[2 * t + 1][p];
      const float py = (float)(y + di - PAD) + dy;
      const float px = (float)(x0 + p + dj - PAD) + dx;
      const float fy = floorf(py), fx = floorf(px);
      const int iy0 = (int)fy, ix0 = (int)fx;
      const int iy1 = iy0 + 1, ix1 = ix0 + 1;
      const float wy1 = py - fy, wy0 = 1.0f - wy1;
      const float wx1 = px - fx, wx0 = 1.0f - wx1;
      const bool vy0 = (iy0 >= 0) && (iy0 < H_);
      const bool vy1 = (iy1 >= 0) && (iy1 < H_);
      const bool vx0 = (ix0 >= 0) && (ix0 < W_);
      const bool vx1 = (ix1 >= 0) && (ix1 < W_);
      const int cy0 = min(max(iy0, 0), H_ - 1);
      const int cy1 = min(max(iy1, 0), H_ - 1);
      const int cx0 = min(max(ix0, 0), W_ - 1);
      const int cx1 = min(max(ix1, 0), W_ - 1);
      const float w00 = wy0 * wx0 * ((vy0 && vx0) ? 1.0f : 0.0f);
      const float w01 = wy0 * wx1 * ((vy0 && vx1) ? 1.0f : 0.0f);
      const float w10 = wy1 * wx0 * ((vy1 && vx0) ? 1.0f : 0.0f);
      const float w11 = wy1 * wx1 * ((vy1 && vx1) ? 1.0f : 0.0f);
      const int o00 = cy0 * W_ + cx0, o01 = cy0 * W_ + cx1;
      const int o10 = cy1 * W_ + cx0, o11 = cy1 * W_ + cx1;
      const float* pc = xb + (size_t)cb * H_ * W_;
      Pack16 pk;
      #pragma unroll
      for (int cc = 0; cc < 16; ++cc) {
        const float* q = pc + (size_t)cc * H_ * W_;
        const float g = w00 * q[o00] + w01 * q[o01] + w10 * q[o10] + w11 * q[o11];
        pk.h[cc] = (_Float16)g;
      }
      *reinterpret_cast<uint4*>(&sB[p][cb])     = pk.q[0];
      *reinterpret_cast<uint4*>(&sB[p][cb + 8]) = pk.q[1];
    }
    __syncthreads();
    #pragma unroll
    for (int kk = 0; kk < 2; ++kk) {
      const v16h b  = load_b(sB, lane, kk);               // shared by both M-tiles
      const v16h a0 = load_a(&sW[mt * 16][0],       lane, kk);
      const v16h a1 = load_a(&sW[(mt + 2) * 16][0], lane, kk);
      acc0.v = __builtin_amdgcn_wmma_f32_16x16x32_f16(
          false, a0, false, b, (short)0, acc0.v, false, false);
      acc1.v = __builtin_amdgcn_wmma_f32_16x16x32_f16(
          false, a1, false, b, (short)0, acc1.v, false, false);
    }
    __syncthreads();
  }

  // ================= epilogue: D tiles (+ b_def) -> global =================
  {
    const int pp     = nt * 16 + (lane & 15);
    const int rowoff = (lane >> 4) << 3;
    #pragma unroll
    for (int r = 0; r < 8; ++r) {
      const int o0 = mt * 16 + r + rowoff;
      const int o1 = (mt + 2) * 16 + r + rowoff;
      out[(((size_t)n * CO_ + o0) * H_ + y) * W_ + x0 + pp] = acc0.f[r] + sBdef[o0];
      out[(((size_t)n * CO_ + o1) * H_ + y) * W_ + x0 + pp] = acc1.f[r] + sBdef[o1];
    }
  }
}

} // namespace

extern "C" void kernel_launch(void* const* d_in, const int* in_sizes, int n_in,
                              void* d_out, int out_size, void* d_ws, size_t ws_size,
                              hipStream_t stream) {
  const float* x     = (const float*)d_in[0];
  const float* w_off = (const float*)d_in[1];
  const float* b_off = (const float*)d_in[2];
  const float* w_def = (const float*)d_in[3];
  const float* b_def = (const float*)d_in[4];
  float* out = (float*)d_out;
  (void)in_sizes; (void)n_in; (void)out_size;

  dim3 grid(W_ / TILE, H_, N_);
  const bool use_ws = (d_ws != nullptr) && (ws_size >= WS_NEEDED);

  if (use_ws) {
    _Float16* ws = (_Float16*)d_ws;
    const int prep_blocks = (WS_DEF_ELEMS + THREADS - 1) / THREADS;  // covers both regions
    prep_weights<<<prep_blocks, dim3(THREADS), 0, stream>>>(w_off, w_def, ws);
    deform_conv_fused<true><<<grid, dim3(THREADS), 0, stream>>>(
        x, w_off, b_off, w_def, b_def, ws, out);
  } else {
    deform_conv_fused<false><<<grid, dim3(THREADS), 0, stream>>>(
        x, w_off, b_off, w_def, b_def, nullptr, out);
  }
}
